// CausalSelfAttention_90340342104487
// MI455X (gfx1250) — matrix operand order repaired and checked by reading, not compiled, not run on hardware
//
#include <hip/hip_runtime.h>
#include <math.h>

// ---------------------------------------------------------------------------
// CDNA5 (gfx1250) implementation of clustered causal self-attention.
// Matrix math via v_wmma_f32_16x16x32_f16 (wave32 WMMA); GEMM tile staging via
// the Tensor Data Mover (tensor_load_to_lds, 6-arg clang-23 form).
// ---------------------------------------------------------------------------

typedef __attribute__((ext_vector_type(16))) _Float16     v16h;
typedef __attribute__((ext_vector_type(8)))  float        v8f;
typedef __attribute__((ext_vector_type(4)))  unsigned int u32x4;
typedef __attribute__((ext_vector_type(8)))  int          i32x8;
typedef __attribute__((ext_vector_type(4)))  int          i32x4;

#define BB   4
#define TT   4096
#define CC   1024
#define HH   16
#define HD   64
#define NC   64
#define GS   256
#define NG   16      // groups = T / GS
#define NCPG 4       // clusters per group
#define CMAX 60      // (NG-1)*NCPG

#ifndef INFINITY
#define INFINITY __builtin_huge_valf()
#endif

#if __has_builtin(__builtin_amdgcn_tensor_load_to_lds)
#define HAS_TDM 1
#else
#define HAS_TDM 0
#endif

// D = A(16x32 f16) x B(32x16 f16) + C(16x16 f32)
__device__ __forceinline__ v8f wmma_f16(v16h a, v16h b, v8f c) {
  return __builtin_amdgcn_wmma_f32_16x16x32_f16(
      /*neg_a=*/false, a, /*neg_b=*/false, b,
      /*c_mod=*/(short)0, c, /*reuse_a=*/false, /*reuse_b=*/false);
}

// ---- fragment loaders per ISA 7.12.2 (wave32) ------------------------------
// A 16x32 f16: lanes 0-15 row=lane, K {0..7,16..23}; lanes 16-31 K {8..15,24..31}
__device__ __forceinline__ v16h frag_a_h(const _Float16* p, int ld) {
  const int lane = threadIdx.x & 31;
  const int row  = lane & 15;
  const int k0   = (lane & 16) ? 8 : 0;
  const _Float16* r = p + row * ld + k0;
  v16h a;
#pragma unroll
  for (int i = 0; i < 8; ++i) a[i] = r[i];
#pragma unroll
  for (int i = 0; i < 8; ++i) a[i + 8] = r[16 + i];
  return a;
}

// Same A layout, f32 source (probability tile in LDS), convert on load.
__device__ __forceinline__ v16h frag_a_f(const float* p, int ld) {
  const int lane = threadIdx.x & 31;
  const int row  = lane & 15;
  const int k0   = (lane & 16) ? 8 : 0;
  const float* r = p + row * ld + k0;
  v16h a;
#pragma unroll
  for (int i = 0; i < 8; ++i) a[i] = (_Float16)r[i];
#pragma unroll
  for (int i = 0; i < 8; ++i) a[i + 8] = (_Float16)r[16 + i];
  return a;
}

// B 32x16 f16, storage [n][k] (n-major; contiguous K per n).
__device__ __forceinline__ v16h frag_b_nmajor(const _Float16* p, int ld) {
  const int lane = threadIdx.x & 31;
  const int n  = lane & 15;
  const int k0 = (lane & 16) ? 8 : 0;
  const _Float16* r = p + n * ld + k0;
  v16h b;
#pragma unroll
  for (int i = 0; i < 8; ++i) b[i] = r[i];
#pragma unroll
  for (int i = 0; i < 8; ++i) b[i + 8] = r[16 + i];
  return b;
}

// B 32x16 f16, storage [k][n] (k-major; e.g. V stored [key][d], n = d).
__device__ __forceinline__ v16h frag_b_kmajor(const _Float16* p, int ld) {
  const int lane = threadIdx.x & 31;
  const int n  = lane & 15;
  const int k0 = (lane & 16) ? 8 : 0;
  v16h b;
#pragma unroll
  for (int i = 0; i < 16; ++i) b[i] = p[(k0 + (i < 8 ? i : i + 8)) * ld + n];
  return b;
}

// Sinusoidal PE matching reference: S[p,2u]=sin(p*f_u), S[p,2u+1]=cos(p*f_u),
// f_u = 10000^{-u/32}.
__device__ __forceinline__ float pe_val(float pos, int m) {
  const int u = m >> 1;
  const float inv = __expf(-0.28782313662425575f * (float)u); // ln(1e4)/32
  const float a = pos * inv;
  return (m & 1) ? __cosf(a) : __sinf(a);
}

#if HAS_TDM
// ---------------------------------------------------------------------------
// Tensor Data Mover: 2D tile of 2-byte elements -> LDS, with LDS padding of
// 4 DWORDs every 16 DWORDs (64B data row -> 80B LDS stride = [rows][40] f16).
// D# per CDNA5 ISA ch.8: group0 = {count/type/addr}, group1 = dims/strides.
// This toolchain's builtin is the 6-arg clang-23 form:
//   (u32x4 g0, i32x8 g1, i32x4 g2, i32x4 g3, i32x8 pad, i32 cpol)
// ---------------------------------------------------------------------------
__device__ __forceinline__ void tdm_load_2d_f16(unsigned lds_off,
                                                const void* gptr,
                                                int tensor_w, int tensor_h,
                                                int tile_w, int tile_h,
                                                int row_stride) {
  const unsigned long long ga = (unsigned long long)gptr;
  u32x4 g0;
  g0[0] = 1u;                                   // count=1, is_restore=0
  g0[1] = lds_off;                              // lds_addr (bytes)
  g0[2] = (unsigned)(ga & 0xFFFFFFFFull);       // global_addr[31:0]
  g0[3] = (unsigned)((ga >> 32) & 0x01FFFFFFull) | (2u << 30);  // addr[56:32]|type=2
  i32x8 g1;
  g1[0] = (1 << 16)    // data_size = 1 (2 bytes)
        | (1 << 20)    // pad_enable
        | (3 << 22)    // pad_interval: 16 DWORDs
        | (3 << 25);   // pad_amount:   4 DWORDs
  g1[1] = (tensor_w & 0xFFFF) << 16;                                // tensor_dim0 lo
  g1[2] = ((unsigned)tensor_w >> 16) | ((tensor_h & 0xFFFF) << 16); // dim0 hi | dim1 lo
  g1[3] = ((unsigned)tensor_h >> 16) | ((tile_w & 0xFFFF) << 16);   // dim1 hi | tile_dim0
  g1[4] = (tile_h & 0xFFFF);                                        // tile_dim1 (tile_dim2=0)
  g1[5] = row_stride;                                               // tensor_dim0_stride lo
  g1[6] = 0;
  g1[7] = 0;
  const i32x4 z4 = {0, 0, 0, 0};
  const i32x8 z8 = {0, 0, 0, 0, 0, 0, 0, 0};
  __builtin_amdgcn_tensor_load_to_lds(g0, g1, z4, z4, z8, 0);
}
#endif

// ---------------------------------------------------------------------------
// Kernel 1: f32 -> f16 elementwise convert (x, w_attn, w_proj)
// ---------------------------------------------------------------------------
__global__ __launch_bounds__(256) void cvt_f16_kernel(const float* __restrict__ s,
                                                      _Float16* __restrict__ d,
                                                      size_t n) {
  size_t i = (size_t)blockIdx.x * 256 + threadIdx.x;
  if (i < n) d[i] = (_Float16)s[i];
}

// ---------------------------------------------------------------------------
// Kernel 2/7: WMMA GEMM  C[M,N] = A[M,K](f16) * Bw[N,K]^T(f16) + bias[N]
// block = 256 thr (8 waves), tile 64(M) x 128(N), K step 32.
// Tiles staged into LDS by the TDM (one wave issues both descriptors, waits
// TENSORcnt, barrier publishes); falls back to vector copies without TDM.
// ---------------------------------------------------------------------------
template <typename OutT>
__global__ __launch_bounds__(256) void gemm_kernel(
    const _Float16* __restrict__ A, const _Float16* __restrict__ Bw,
    const float* __restrict__ bias, OutT* __restrict__ C,
    int M, int N, int K) {
  __shared__ __align__(16) _Float16 As[64][40];
  __shared__ __align__(16) _Float16 Bs[128][40];
  const int tid  = threadIdx.x;
  const int wid  = tid >> 5;
  const int lane = tid & 31;
  const int wm = wid >> 2;          // 0..1
  const int wn = wid & 3;           // 0..3
  const int mbase = blockIdx.x * 64;
  const int nbase = blockIdx.y * 128;

  v8f acc00 = {}, acc01 = {}, acc10 = {}, acc11 = {};

#if !HAS_TDM
  const int arow = tid >> 2, acg = (tid & 3) * 8;   // A: 8 halves / thread
  const int brow = tid >> 1, bcg = (tid & 1) * 16;  // B: 16 halves / thread
#endif

  for (int k0 = 0; k0 < K; k0 += 32) {
#if HAS_TDM
    if (tid == 0) {
      tdm_load_2d_f16((unsigned)(size_t)(void*)&As[0][0],
                      A + (size_t)mbase * K + k0, K, M, 32, 64, K);
      tdm_load_2d_f16((unsigned)(size_t)(void*)&Bs[0][0],
                      Bw + (size_t)nbase * K + k0, K, N, 32, 128, K);
      __builtin_amdgcn_s_wait_tensorcnt((short)0);
    }
#else
    *(uint4*)(&As[arow][acg]) =
        *(const uint4*)(A + (size_t)(mbase + arow) * K + k0 + acg);
    const _Float16* bsrc = Bw + (size_t)(nbase + brow) * K + k0 + bcg;
    *(uint4*)(&Bs[brow][bcg])     = *(const uint4*)(bsrc);
    *(uint4*)(&Bs[brow][bcg + 8]) = *(const uint4*)(bsrc + 8);
#endif
    __syncthreads();

    v16h a0 = frag_a_h(&As[wm * 32][0], 40);
    v16h a1 = frag_a_h(&As[wm * 32 + 16][0], 40);
    v16h b0 = frag_b_nmajor(&Bs[wn * 32][0], 40);
    v16h b1 = frag_b_nmajor(&Bs[wn * 32 + 16][0], 40);
    acc00 = wmma_f16(a0, b0, acc00);
    acc01 = wmma_f16(a0, b1, acc01);
    acc10 = wmma_f16(a1, b0, acc10);
    acc11 = wmma_f16(a1, b1, acc11);
    __syncthreads();
  }

  const int col = lane & 15;
  const int rb  = (lane >> 4) * 8;   // C frag: row = r + 8*(lane/16), col = lane%16
#pragma unroll
  for (int ti = 0; ti < 2; ++ti) {
#pragma unroll
    for (int tj = 0; tj < 2; ++tj) {
      v8f a = (ti == 0) ? ((tj == 0) ? acc00 : acc01)
                        : ((tj == 0) ? acc10 : acc11);
      const int n = nbase + wn * 32 + tj * 16 + col;
      const float bv = bias[n];
#pragma unroll
      for (int r = 0; r < 8; ++r) {
        const int m = mbase + wm * 32 + ti * 16 + rb + r;
        C[(size_t)m * N + n] = (OutT)(a[r] + bv);
      }
    }
  }
}

// ---------------------------------------------------------------------------
// Kernel 3: cluster assignment softmax + weighted k/v summaries.
// one block per (b, cluster).
// ---------------------------------------------------------------------------
__global__ __launch_bounds__(256) void cluster_kernel(
    const float* __restrict__ x, const float* __restrict__ w_assign,
    const _Float16* __restrict__ qkv, float* __restrict__ ck,
    float* __restrict__ cv) {
  const int b = blockIdx.x >> 6;
  const int c = blockIdx.x & 63;
  __shared__ float part[64][4];
  __shared__ float wgt[64];
  const int tid = threadIdx.x;
  const int t = tid >> 2, pp = tid & 3;

  const float* xp = x + (((size_t)b * TT) + (size_t)c * 64 + t) * CC + pp * 256;
  const float* wp = w_assign + (size_t)c * CC + pp * 256;
  float s = 0.f;
  for (int i = 0; i < 256; ++i) s += xp[i] * wp[i];
  part[t][pp] = s;
  __syncthreads();
  if (tid < 64) wgt[tid] = part[tid][0] + part[tid][1] + part[tid][2] + part[tid][3];
  __syncthreads();
  if (tid == 0) {
    float m = -INFINITY;
    for (int i = 0; i < 64; ++i) m = fmaxf(m, wgt[i]);
    float sum = 0.f;
    for (int i = 0; i < 64; ++i) { float e = __expf(wgt[i] - m); wgt[i] = e; sum += e; }
    float inv = 1.f / sum;
    for (int i = 0; i < 64; ++i) wgt[i] *= inv;
  }
  __syncthreads();

  const _Float16* base = qkv + (((size_t)b * TT) + (size_t)c * 64) * 3072;
#pragma unroll
  for (int j = 0; j < 4; ++j) {
    const int d = tid + j * 256;
    float ak = 0.f, av = 0.f;
    for (int t2 = 0; t2 < 64; ++t2) {
      const float w = wgt[t2];
      ak += w * (float)base[(size_t)t2 * 3072 + 1024 + d];
      av += w * (float)base[(size_t)t2 * 3072 + 2048 + d];
    }
    ck[(((size_t)b * NC) + c) * CC + d] = ak;
    cv[(((size_t)b * NC) + c) * CC + d] = av;
  }
}

// ---------------------------------------------------------------------------
// Kernel 4: rotary prep for q (pos = i in group, folded 1/sqrt(hd)) and local
// k (pos = g*4 + i).  Reference _rotary: out[2j] = -x[2j+1]*pe[32+j],
// out[2j+1] = x[2j]*pe[j].
// ---------------------------------------------------------------------------
__global__ __launch_bounds__(256) void rope_prep_kernel(
    const _Float16* __restrict__ qkv, _Float16* __restrict__ qh,
    _Float16* __restrict__ klh) {
  const size_t idx = (size_t)blockIdx.x * 256 + threadIdx.x;  // B*T*512 pairs
  const size_t tok = idx >> 9;
  const int p = (int)(idx & 511);
  const int h = p >> 5, j = p & 31;
  const int t = (int)(tok & (TT - 1));
  const int i = t & (GS - 1);
  const int g = (t >> 8) & (NG - 1);

  const size_t base = tok * 3072 + (size_t)h * HD + 2 * j;
  const float q0 = (float)qkv[base],        q1 = (float)qkv[base + 1];
  const float k0 = (float)qkv[base + 1024], k1 = (float)qkv[base + 1025];
  const float qpos = (float)i;
  const float kpos = (float)(g * NCPG + i);
  const float scale = 0.125f;  // 1/sqrt(64), folded into rotated q

  const size_t ob = tok * CC + (size_t)h * HD + 2 * j;
  qh[ob]      = (_Float16)(-q1 * pe_val(qpos, 32 + j) * scale);
  qh[ob + 1]  = (_Float16)( q0 * pe_val(qpos, j)      * scale);
  klh[ob]     = (_Float16)(-k1 * pe_val(kpos, 32 + j));
  klh[ob + 1] = (_Float16)( k0 * pe_val(kpos, j));
}

// ---------------------------------------------------------------------------
// Kernel 5: rotary + f16 convert for cluster k/v; pad clusters 60..63 w/ zero.
// ---------------------------------------------------------------------------
__global__ __launch_bounds__(256) void cluster_prep_kernel(
    const float* __restrict__ ck, const float* __restrict__ cv,
    _Float16* __restrict__ kch, _Float16* __restrict__ vch) {
  const size_t idx = (size_t)blockIdx.x * 256 + threadIdx.x;  // B*64*512 pairs
  const size_t row = idx >> 9;   // b*64 + c
  const int c = (int)(row & 63);
  const int p = (int)(idx & 511);
  const int h = p >> 5, j = p & 31;
  const size_t ob = row * CC + (size_t)h * HD + 2 * j;
  if (c >= CMAX) {
    kch[ob] = (_Float16)0.f; kch[ob + 1] = (_Float16)0.f;
    vch[ob] = (_Float16)0.f; vch[ob + 1] = (_Float16)0.f;
    return;
  }
  const float k0 = ck[ob], k1 = ck[ob + 1];
  const float pos = (float)c;
  kch[ob]     = (_Float16)(-k1 * pe_val(pos, 32 + j));
  kch[ob + 1] = (_Float16)( k0 * pe_val(pos, j));
  vch[ob]     = (_Float16)cv[ob];
  vch[ob + 1] = (_Float16)cv[ob + 1];
}

// ---------------------------------------------------------------------------
// Kernel 6: attention.  grid = B*G*H*8 blocks; 256 thr (8 waves); 32 q-rows
// per block.  Keys: cols 0..63 = cluster (60 valid + pad), 64..319 = local.
// Score/softmax tile in LDS (f32); K/V fragments stream from L2 (192 MB).
// ---------------------------------------------------------------------------
#define SLD 324
__global__ __launch_bounds__(256) void attn_kernel(
    const _Float16* __restrict__ qh, const _Float16* __restrict__ klh,
    const _Float16* __restrict__ qkv, const _Float16* __restrict__ kch,
    const _Float16* __restrict__ vch, _Float16* __restrict__ yh) {
  __shared__ float S[32][SLD];

  const int idx = blockIdx.x;
  const int qb = idx & 7;
  const int h  = (idx >> 3) & 15;
  const int g  = (idx >> 7) & 15;
  const int b  = idx >> 11;
  const int tid = threadIdx.x, lane = tid & 31, wid = tid >> 5;
  const size_t tokbase = (size_t)b * TT + (size_t)g * GS;
  const int qrow0 = qb * 32;

  // ---- scores ----
  {
    const int rt = wid & 1;
    const _Float16* qp = qh + (tokbase + qrow0 + rt * 16) * CC + h * HD;
    const v16h aq0 = frag_a_h(qp, CC);        // d 0..31
    const v16h aq1 = frag_a_h(qp + 32, CC);   // d 32..63
    const int col = lane & 15;
    const int rb  = (lane >> 4) * 8;
#pragma unroll
    for (int i = 0; i < 5; ++i) {
      const int kt = (wid >> 1) + 4 * i;      // 0..19
      v16h b0, b1;
      if (kt < 4) {
        const _Float16* kp = kch + ((size_t)b * NC + kt * 16) * CC + h * HD;
        b0 = frag_b_nmajor(kp, CC);
        b1 = frag_b_nmajor(kp + 32, CC);
      } else {
        const _Float16* kp = klh + (tokbase + (kt - 4) * 16) * CC + h * HD;
        b0 = frag_b_nmajor(kp, CC);
        b1 = frag_b_nmajor(kp + 32, CC);
      }
      v8f c = {};
      c = wmma_f16(aq0, b0, c);
      c = wmma_f16(aq1, b1, c);
      const int kcol = kt * 16 + col;
#pragma unroll
      for (int r = 0; r < 8; ++r) {
        const int qrow = qrow0 + rt * 16 + rb + r;   // position in group
        bool ok;
        if (kt < 4) ok = (kcol < g * NCPG);          // cluster strictly before group
        else        ok = ((kcol - 64) <= qrow);      // causal within group
        S[rt * 16 + rb + r][kcol] = ok ? c[r] : -INFINITY;
      }
    }
  }
  __syncthreads();

  // ---- row softmax over 320 keys (f32) ----
  if (tid < 32) {
    float* row = &S[tid][0];
    float m = -INFINITY;
    for (int i = 0; i < 320; ++i) m = fmaxf(m, row[i]);
    float sum = 0.f;
    for (int i = 0; i < 320; ++i) { float e = __expf(row[i] - m); row[i] = e; sum += e; }
    const float inv = 1.f / sum;
    for (int i = 0; i < 320; ++i) row[i] *= inv;
  }
  __syncthreads();

  // ---- PV: wave -> (row tile rt2, d tile dt); K dim = 320 keys, step 32 ----
  {
    const int rt2 = wid & 1, dt = wid >> 1;
    v8f acc = {};
#pragma unroll
    for (int ks = 0; ks < 10; ++ks) {
      const v16h ap = frag_a_f(&S[rt2 * 16][ks * 32], SLD);
      v16h bv;
      if (ks < 2) {
        const _Float16* vp = vch + ((size_t)b * NC + ks * 32) * CC + h * HD + dt * 16;
        bv = frag_b_kmajor(vp, CC);
      } else {
        const _Float16* vp =
            qkv + (tokbase + (ks - 2) * 32) * 3072 + 2048 + h * HD + dt * 16;
        bv = frag_b_kmajor(vp, 3072);
      }
      acc = wmma_f16(ap, bv, acc);
    }
    const int col = lane & 15;
    const int rb  = (lane >> 4) * 8;
#pragma unroll
    for (int r = 0; r < 8; ++r) {
      const int qrow = qrow0 + rt2 * 16 + rb + r;
      yh[(tokbase + qrow) * CC + h * HD + dt * 16 + col] = (_Float16)acc[r];
    }
  }
}

// ---------------------------------------------------------------------------
// Workspace layout (bytes; all 256-aligned):
//   xh   @ 0          (33,554,432)  f16 x
//   qkv  @ 33554432   (100,663,296) f16 [B,T,3C]
//   qh   @ 134217728  (33,554,432)  f16 rotated+scaled q
//   klh  @ 167772160  (33,554,432)  f16 rotated local k
//   ck   @ 201326592  (1,048,576)   f32 cluster k
//   cv   @ 202375168  (1,048,576)   f32 cluster v
//   kch  @ 203423744  (524,288)     f16 rotated cluster k (64 rows, padded)
//   vch  @ 203948032  (524,288)     f16 cluster v (padded)
//   yh   @ 204472320  (33,554,432)  f16 attention output
//   wah  @ 238026752  (6,291,456)   f16 w_attn
//   wph  @ 244318208  (2,097,152)   f16 w_proj
//   total ~235 MB
// ---------------------------------------------------------------------------
extern "C" void kernel_launch(void* const* d_in, const int* in_sizes, int n_in,
                              void* d_out, int out_size, void* d_ws, size_t ws_size,
                              hipStream_t stream) {
  const float* x        = (const float*)d_in[0];
  const float* w_attn   = (const float*)d_in[1];
  const float* b_attn   = (const float*)d_in[2];
  const float* w_proj   = (const float*)d_in[3];
  const float* b_proj   = (const float*)d_in[4];
  const float* w_assign = (const float*)d_in[5];
  float* out = (float*)d_out;

  char* ws = (char*)d_ws;
  _Float16* xh  = (_Float16*)(ws + 0ULL);
  _Float16* qkv = (_Float16*)(ws + 33554432ULL);
  _Float16* qh  = (_Float16*)(ws + 134217728ULL);
  _Float16* klh = (_Float16*)(ws + 167772160ULL);
  float*    ck  = (float*)(ws + 201326592ULL);
  float*    cv  = (float*)(ws + 202375168ULL);
  _Float16* kch = (_Float16*)(ws + 203423744ULL);
  _Float16* vch = (_Float16*)(ws + 203948032ULL);
  _Float16* yh  = (_Float16*)(ws + 204472320ULL);
  _Float16* wah = (_Float16*)(ws + 238026752ULL);
  _Float16* wph = (_Float16*)(ws + 244318208ULL);

  // 1) f32 -> f16 conversions (x and both weight matrices, converted once)
  cvt_f16_kernel<<<65536, 256, 0, stream>>>(x, xh, (size_t)BB * TT * CC);
  cvt_f16_kernel<<<12288, 256, 0, stream>>>(w_attn, wah, (size_t)3 * CC * CC);
  cvt_f16_kernel<<<4096, 256, 0, stream>>>(w_proj, wph, (size_t)CC * CC);

  // 2) qkv = xh @ w_attn^T + b_attn   (M=16384, N=3072, K=1024), f16 out
  gemm_kernel<_Float16><<<dim3(256, 24), 256, 0, stream>>>(
      xh, wah, b_attn, qkv, BB * TT, 3 * CC, CC);

  // 3) cluster softmax + summaries
  cluster_kernel<<<BB * NC, 256, 0, stream>>>(x, w_assign, qkv, ck, cv);

  // 4) rotary prep for q / local k
  rope_prep_kernel<<<32768, 256, 0, stream>>>(qkv, qh, klh);

  // 5) rotary prep for cluster k/v (+ zero pad rows 60..63)
  cluster_prep_kernel<<<512, 256, 0, stream>>>(ck, cv, kch, vch);

  // 6) attention -> yh (f16)
  attn_kernel<<<BB * NG * HH * 8, 256, 0, stream>>>(qh, klh, qkv, kch, vch, yh);

  // 7) out = yh @ w_proj^T + b_proj   (M=16384, N=1024, K=1024), f32 out
  gemm_kernel<float><<<dim3(256, 8), 256, 0, stream>>>(
      yh, wph, b_proj, out, BB * TT, CC, CC);
}
